// Net_3874060501607
// MI455X (gfx1250) — matrix operand order repaired and checked by reading, not compiled, not run on hardware
//
#include <hip/hip_runtime.h>
#include <math.h>

// ---------------------------------------------------------------------------
// GNN forward: GraphConv(norm=both) -> SAGEConv(mean) -> Dense, all with ELU.
// N=50000, E=800000, D_IN=128, D1=150(->160), D2=100(->128), D_OUT=64.
// GEMMs: v_wmma_f32_16x16x32_bf16; 80x16 output strip per wave (MB=5 divides
// MT=3125 exactly -> guard-free main loop); weights prepacked in B-fragment
// order (one contiguous 32B load per fragment); scalar (SGPR) tile indices.
// Scatters: wave-per-edge float4 gathers + global f32 atomics (L2-resident).
// ---------------------------------------------------------------------------

typedef __attribute__((ext_vector_type(16))) __bf16 v16bf;
typedef __attribute__((ext_vector_type(8)))  float  v8f;

#define D_IN   128
#define D1P    160   // 150 padded to 16*10
#define D2P    128   // 100 padded to 16*8
#define D_OUT  64
#define MB     5     // M-tiles per wave; 3125 % 5 == 0 -> no tail for N=50000

__device__ __forceinline__ float eluf(float x) {
    return x > 0.f ? x : (__expf(x) - 1.f);
}

__device__ __forceinline__ v8f wmma_bf16(v16bf a, v16bf b, v8f c) {
    return __builtin_amdgcn_wmma_f32_16x16x32_bf16(false, a, false, b,
                                                   (short)0, c, false, false);
}

// A fragment: 16x32 bf16 tile from row-major f32 matrix, with per-row scale.
__device__ __forceinline__ v16bf load_a_f32(const float* __restrict__ A, int lda,
                                            int row0, int k0, int lane, float scale) {
    int m  = lane & 15;
    int kb = (lane >> 4) << 3;
    const float* ar = A + (long)(row0 + m) * lda + k0 + kb;
    v16bf a;
#pragma unroll
    for (int e = 0; e < 8; ++e) a[e]     = (__bf16)(ar[e] * scale);
#pragma unroll
    for (int e = 0; e < 8; ++e) a[8 + e] = (__bf16)(ar[16 + e] * scale);
    return a;
}

// B fragment from prepacked weights: tile-major, then lane, then 16 bf16.
__device__ __forceinline__ v16bf load_b_packed(const __bf16* __restrict__ Bp,
                                               int NT, int kt, int nt, int lane) {
    const v16bf* p = (const v16bf*)Bp;
    return p[((long)(kt * NT + nt)) * 32 + lane];
}

// ---------------------------------------------------------------------------

__global__ void zero4_kernel(float4* __restrict__ p, long n4) {
    long i = (long)blockIdx.x * blockDim.x + threadIdx.x;
    long stride = (long)gridDim.x * blockDim.x;
    float4 z = make_float4(0.f, 0.f, 0.f, 0.f);
    for (; i < n4; i += stride) p[i] = z;
}

// f32 weight [rows][cols] -> zero-padded bf16 in WMMA B-fragment order.
__global__ void packB_kernel(const float* __restrict__ W, int rows, int cols,
                             __bf16* __restrict__ out, int NT, int total) {
    int i = blockIdx.x * blockDim.x + threadIdx.x;
    if (i >= total) return;
    int e    = i & 15;
    int lane = (i >> 4) & 31;
    int tile = i >> 9;
    int nt = tile % NT, kt = tile / NT;
    int n = nt * 16 + (lane & 15);
    int k = kt * 32 + ((lane >> 4) << 3) + (e & 7) + ((e >> 3) << 4);
    float v = (k < rows && n < cols) ? W[k * cols + n] : 0.f;
    out[i] = (__bf16)v;
}

__global__ void padbias_kernel(const float* __restrict__ b, int n,
                               float* __restrict__ out, int pn) {
    int i = blockIdx.x * blockDim.x + threadIdx.x;
    if (i < pn) out[i] = (i < n) ? b[i] : 0.f;
}

__global__ void degree_kernel(const int* __restrict__ src, const int* __restrict__ dst,
                              float* __restrict__ deg_out, float* __restrict__ deg_in, int E) {
    int i = blockIdx.x * blockDim.x + threadIdx.x;
    if (i < E) {
        atomicAdd(&deg_out[src[i]], 1.f);
        atomicAdd(&deg_in[dst[i]], 1.f);
    }
}

// agg[dst] += x[src] * rsqrt(max(deg_out[src],1)); one wave per edge.
__global__ void scatter1_kernel(const float* __restrict__ x,
                                const int* __restrict__ src, const int* __restrict__ dst,
                                const float* __restrict__ deg_out,
                                float* __restrict__ agg, int E) {
    int gid  = blockIdx.x * blockDim.x + threadIdx.x;
    int edge = __builtin_amdgcn_readfirstlane(gid >> 5);   // scalar edge index
    int lane = threadIdx.x & 31;
    if (edge >= E) return;
    int s = src[edge], d = dst[edge];                      // scalar loads
    float norm = rsqrtf(fmaxf(deg_out[s], 1.f));
    float4 v = ((const float4*)(x + (long)s * D_IN))[lane];  // 32*4 = 128 cols
    float* ad = agg + (long)d * D_IN + lane * 4;
    atomicAdd(ad + 0, v.x * norm);
    atomicAdd(ad + 1, v.y * norm);
    atomicAdd(ad + 2, v.z * norm);
    atomicAdd(ad + 3, v.w * norm);
}

// neigh[dst] += h1[src] over 160 padded cols (pad cols zero in h1).
__global__ void scatter2_kernel(const float* __restrict__ h1,
                                const int* __restrict__ src, const int* __restrict__ dst,
                                float* __restrict__ neigh, int E) {
    int gid  = blockIdx.x * blockDim.x + threadIdx.x;
    int edge = __builtin_amdgcn_readfirstlane(gid >> 5);
    int lane = threadIdx.x & 31;
    if (edge >= E) return;
    int s = src[edge], d = dst[edge];
    const float4* hs = (const float4*)(h1 + (long)s * D1P);
    float* nd = neigh + (long)d * D1P;
#pragma unroll
    for (int it = 0; it < 2; ++it) {
        int c4 = lane + it * 32;        // 40 float4 chunks cover 160 cols
        if (c4 < D1P / 4) {
            float4 v = hs[c4];
            atomicAdd(nd + c4 * 4 + 0, v.x);
            atomicAdd(nd + c4 * 4 + 1, v.y);
            atomicAdd(nd + c4 * 4 + 2, v.z);
            atomicAdd(nd + c4 * 4 + 3, v.w);
        }
    }
}

// ---------------------------------------------------------------------------
// GEMM bodies (templated on strip height so the tail path reuses code).
// mt0/nt are SGPR values (derived from readfirstlane'd wave id).
// ---------------------------------------------------------------------------

template <int MBv>
__device__ __forceinline__ void gemm1_body(const float* __restrict__ agg,
                                           const float* __restrict__ deg_in,
                                           const __bf16* __restrict__ Bp,
                                           const float* __restrict__ bias,
                                           float* __restrict__ h1,
                                           int mt0, int nt, int lane) {
    const int NT = D1P / 16;
    float scale[MBv];
#pragma unroll
    for (int i = 0; i < MBv; ++i)
        scale[i] = rsqrtf(fmaxf(deg_in[(mt0 + i) * 16 + (lane & 15)], 1.f));

    v8f c[MBv] = {};
#pragma unroll 1
    for (int k0 = 0; k0 < D_IN; k0 += 32) {
        v16bf b = load_b_packed(Bp, NT, k0 >> 5, nt, lane);
#pragma unroll
        for (int i = 0; i < MBv; ++i)
            c[i] = wmma_bf16(load_a_f32(agg, D_IN, (mt0 + i) * 16, k0, lane, scale[i]),
                             b, c[i]);
    }
    int nc = nt * 16 + (lane & 15);
    float bv = bias[nc];
#pragma unroll
    for (int i = 0; i < MBv; ++i) {
        int mbase = (mt0 + i) * 16 + ((lane >> 4) << 3);
#pragma unroll
        for (int r = 0; r < 8; ++r)
            h1[(long)(mbase + r) * D1P + nc] = eluf(c[i][r] + bv);
    }
}

template <int MBv>
__device__ __forceinline__ void gemm2_body(const float* __restrict__ h1,
                                           const float* __restrict__ neigh,
                                           const float* __restrict__ deg_in,
                                           const __bf16* __restrict__ Bsp,
                                           const __bf16* __restrict__ Bnp,
                                           const float* __restrict__ bias,
                                           float* __restrict__ h2,
                                           int mt0, int nt, int lane) {
    const int NT = D2P / 16;
    v8f c[MBv] = {};
#pragma unroll 1
    for (int k0 = 0; k0 < D1P; k0 += 32) {          // self path
        v16bf bs = load_b_packed(Bsp, NT, k0 >> 5, nt, lane);
#pragma unroll
        for (int i = 0; i < MBv; ++i)
            c[i] = wmma_bf16(load_a_f32(h1, D1P, (mt0 + i) * 16, k0, lane, 1.f),
                             bs, c[i]);
    }
    float rdeg[MBv];
#pragma unroll
    for (int i = 0; i < MBv; ++i)
        rdeg[i] = 1.f / fmaxf(deg_in[(mt0 + i) * 16 + (lane & 15)], 1.f);
#pragma unroll 1
    for (int k0 = 0; k0 < D1P; k0 += 32) {          // neighbor-mean path
        v16bf bn = load_b_packed(Bnp, NT, k0 >> 5, nt, lane);
#pragma unroll
        for (int i = 0; i < MBv; ++i)
            c[i] = wmma_bf16(load_a_f32(neigh, D1P, (mt0 + i) * 16, k0, lane, rdeg[i]),
                             bn, c[i]);
    }
    int nc = nt * 16 + (lane & 15);
    float bv = bias[nc];
#pragma unroll
    for (int i = 0; i < MBv; ++i) {
        int mbase = (mt0 + i) * 16 + ((lane >> 4) << 3);
#pragma unroll
        for (int r = 0; r < 8; ++r)
            h2[(long)(mbase + r) * D2P + nc] = eluf(c[i][r] + bv);
    }
}

template <int MBv>
__device__ __forceinline__ void gemm3_body(const float* __restrict__ h2,
                                           const __bf16* __restrict__ Bp,
                                           const float* __restrict__ bias,
                                           float* __restrict__ out,
                                           int mt0, int nt, int lane) {
    const int NT = D_OUT / 16;
    v8f c[MBv] = {};
#pragma unroll 1
    for (int k0 = 0; k0 < D2P; k0 += 32) {
        v16bf b = load_b_packed(Bp, NT, k0 >> 5, nt, lane);
#pragma unroll
        for (int i = 0; i < MBv; ++i)
            c[i] = wmma_bf16(load_a_f32(h2, D2P, (mt0 + i) * 16, k0, lane, 1.f),
                             b, c[i]);
    }
    int nc = nt * 16 + (lane & 15);
    float bv = bias[nc];
#pragma unroll
    for (int i = 0; i < MBv; ++i) {
        int mbase = (mt0 + i) * 16 + ((lane >> 4) << 3);
#pragma unroll
        for (int r = 0; r < 8; ++r)
            out[(long)(mbase + r) * D_OUT + nc] = eluf(c[i][r] + bv);
    }
}

// ---------------------------------------------------------------------------

__global__ void gemm1_kernel(const float* __restrict__ agg, const float* __restrict__ deg_in,
                             const __bf16* __restrict__ Bp, const float* __restrict__ bias,
                             float* __restrict__ h1, int N) {
    int lane = threadIdx.x & 31;
    int wid  = __builtin_amdgcn_readfirstlane((blockIdx.x * blockDim.x + threadIdx.x) >> 5);
    const int NT = D1P / 16;
    int MT = N / 16, strips = MT / MB, mainw = strips * NT;
    if (wid < mainw) {
        gemm1_body<MB>(agg, deg_in, Bp, bias, h1, (wid / NT) * MB, wid % NT, lane);
    } else {
        int w2 = wid - mainw;                       // tail (empty for N=50000)
        if (w2 < (MT - strips * MB) * NT)
            gemm1_body<1>(agg, deg_in, Bp, bias, h1, strips * MB + w2 / NT, w2 % NT, lane);
    }
}

__global__ void gemm2_kernel(const float* __restrict__ h1, const float* __restrict__ neigh,
                             const float* __restrict__ deg_in,
                             const __bf16* __restrict__ Bsp, const __bf16* __restrict__ Bnp,
                             const float* __restrict__ bias,
                             float* __restrict__ h2, int N) {
    int lane = threadIdx.x & 31;
    int wid  = __builtin_amdgcn_readfirstlane((blockIdx.x * blockDim.x + threadIdx.x) >> 5);
    const int NT = D2P / 16;
    int MT = N / 16, strips = MT / MB, mainw = strips * NT;
    if (wid < mainw) {
        gemm2_body<MB>(h1, neigh, deg_in, Bsp, Bnp, bias, h2, (wid / NT) * MB, wid % NT, lane);
    } else {
        int w2 = wid - mainw;
        if (w2 < (MT - strips * MB) * NT)
            gemm2_body<1>(h1, neigh, deg_in, Bsp, Bnp, bias, h2,
                          strips * MB + w2 / NT, w2 % NT, lane);
    }
}

__global__ void gemm3_kernel(const float* __restrict__ h2,
                             const __bf16* __restrict__ Bp, const float* __restrict__ bias,
                             float* __restrict__ out, int N) {
    int lane = threadIdx.x & 31;
    int wid  = __builtin_amdgcn_readfirstlane((blockIdx.x * blockDim.x + threadIdx.x) >> 5);
    const int NT = D_OUT / 16;
    int MT = N / 16, strips = MT / MB, mainw = strips * NT;
    if (wid < mainw) {
        gemm3_body<MB>(h2, Bp, bias, out, (wid / NT) * MB, wid % NT, lane);
    } else {
        int w2 = wid - mainw;
        if (w2 < (MT - strips * MB) * NT)
            gemm3_body<1>(h2, Bp, bias, out, strips * MB + w2 / NT, w2 % NT, lane);
    }
}

// ---------------------------------------------------------------------------

extern "C" void kernel_launch(void* const* d_in, const int* in_sizes, int n_in,
                              void* d_out, int out_size, void* d_ws, size_t ws_size,
                              hipStream_t stream) {
    const float* x  = (const float*)d_in[0];
    const int*   src = (const int*)d_in[1];
    const int*   dst = (const int*)d_in[2];
    const float* W1 = (const float*)d_in[3];
    const float* b1 = (const float*)d_in[4];
    const float* Wn = (const float*)d_in[5];
    const float* Ws = (const float*)d_in[6];
    const float* b2 = (const float*)d_in[7];
    const float* W3 = (const float*)d_in[8];
    const float* b3 = (const float*)d_in[9];
    float* out = (float*)d_out;

    const int N = in_sizes[0] / D_IN;   // 50000 (multiple of 16)
    const int E = in_sizes[1];          // 800000

    // ---- workspace carve (f32 region, then 64B-aligned packed-bf16 region) --
    float* f = (float*)d_ws;
    float* deg_out = f;  f += N;
    float* deg_in  = f;  f += N;
    float* agg1    = f;  f += (long)N * D_IN;   // aliased as h2 after GEMM1
    float* h1      = f;  f += (long)N * D1P;
    float* neigh   = f;  f += (long)N * D1P;
    float* b1p     = f;  f += D1P;
    float* b2p     = f;  f += D2P;
    uintptr_t pa = ((uintptr_t)f + 63) & ~(uintptr_t)63;
    __bf16* W1b = (__bf16*)pa;
    __bf16* Wsb = W1b + (D_IN / 32) * (D1P / 16) * 512;
    __bf16* Wnb = Wsb + (D1P / 32) * (D2P / 16) * 512;
    __bf16* W3b = Wnb + (D1P / 32) * (D2P / 16) * 512;
    float* h2 = agg1;                           // reuse: agg1 dead after GEMM1

    const int T = 256;
    const int MT = N / 16;
    const int strips = MT / MB;
    const int tail = MT - strips * MB;          // 0 for N=50000

    // 1) zero accumulators (deg_out|deg_in|agg1 contiguous, then neigh)
    zero4_kernel<<<2048, T, 0, stream>>>((float4*)deg_out, (long)N * (2 + D_IN) / 4);
    zero4_kernel<<<2048, T, 0, stream>>>((float4*)neigh, (long)N * D1P / 4);

    // 2) weights -> packed bf16 B-fragments, biases -> padded f32
    {
        int t1 = (D_IN / 32) * (D1P / 16) * 512;
        int t2 = (D1P / 32) * (D2P / 16) * 512;
        int t3 = (D2P / 32) * (D_OUT / 16) * 512;
        packB_kernel<<<(t1 + T - 1) / T, T, 0, stream>>>(W1, D_IN, 150, W1b, D1P / 16, t1);
        packB_kernel<<<(t2 + T - 1) / T, T, 0, stream>>>(Ws, 150, 100, Wsb, D2P / 16, t2);
        packB_kernel<<<(t2 + T - 1) / T, T, 0, stream>>>(Wn, 150, 100, Wnb, D2P / 16, t2);
        packB_kernel<<<(t3 + T - 1) / T, T, 0, stream>>>(W3, 100, D_OUT, W3b, D_OUT / 16, t3);
    }
    padbias_kernel<<<1, T, 0, stream>>>(b1, 150, b1p, D1P);
    padbias_kernel<<<1, T, 0, stream>>>(b2, 100, b2p, D2P);

    // 3) degrees
    degree_kernel<<<(E + T - 1) / T, T, 0, stream>>>(src, dst, deg_out, deg_in, E);

    // 4) GraphConv scatter: agg1[dst] += x[src]*norm_out[src]
    scatter1_kernel<<<(E * 32 + T - 1) / T, T, 0, stream>>>(x, src, dst, deg_out, agg1, E);

    // 5) GEMM1 + norm_in + ELU -> h1 [N][160]
    {
        int waves = (strips + tail) * (D1P / 16);
        gemm1_kernel<<<(waves + 7) / 8, T, 0, stream>>>(agg1, deg_in, W1b, b1p, h1, N);
    }

    // 6) SAGE scatter: neigh[dst] += h1[src]
    scatter2_kernel<<<(E * 32 + T - 1) / T, T, 0, stream>>>(h1, src, dst, neigh, E);

    // 7) GEMM2 (self + mean-neighbor) + ELU -> h2 [N][128]
    {
        int waves = (strips + tail) * (D2P / 16);
        gemm2_kernel<<<(waves + 7) / 8, T, 0, stream>>>(h1, neigh, deg_in, Wsb, Wnb, b2p, h2, N);
    }

    // 8) GEMM3 + ELU -> out [N][64]
    {
        int waves = (strips + tail) * (D_OUT / 16);
        gemm3_kernel<<<(waves + 7) / 8, T, 0, stream>>>(h2, W3b, b3, out, N);
    }
}
